// Attention_77704548319846
// MI455X (gfx1250) — compile-verified, matrix operand order checked by reading, not run on hardware
//
#include <hip/hip_runtime.h>
#include <hip/hip_bf16.h>

// ---------------------------------------------------------------------------
// Problem dims (fixed by reference)
// ---------------------------------------------------------------------------
constexpr int HD    = 1000;   // hidden size H
constexpr int HD2   = 2000;   // 2H (K of big GEMM)
constexpr int NB    = 64;     // batch
constexpr int SL    = 512;    // source length
constexpr int NPAD  = 1024;   // H padded to 64 n-tiles of 16
constexpr int NTILE = 64;     // n tiles
constexpr int KPAD  = 2016;   // 2H padded to 63 k-steps of 32
constexpr int KSTEPS = 63;
constexpr int MROWS = 64;     // context rows per workgroup
constexpr int LDSROW = 2024;  // bf16 elems per LDS row (pad -> conflict-free b128)

typedef __attribute__((ext_vector_type(16))) __bf16 bf16x16;
typedef __attribute__((ext_vector_type(8)))  __bf16 bf16x8;
typedef __attribute__((ext_vector_type(4)))  __bf16 bf16x4;
typedef __attribute__((ext_vector_type(8)))  float  f32x8;
typedef __attribute__((ext_vector_type(4)))  float  f32x4;

__device__ __forceinline__ f32x8 wmma_bf16(bf16x16 a, bf16x16 b, f32x8 c) {
  // D = A(16x32 bf16) x B(32x16 bf16) + C(16x16 f32)
  return __builtin_amdgcn_wmma_f32_16x16x32_bf16(
      /*neg_a=*/false, a, /*neg_b=*/false, b,
      /*c_mod=*/(short)0, c, /*reuse_a=*/false, /*reuse_b=*/false);
}

__device__ __forceinline__ f32x8 zero8() {
  f32x8 z;
#pragma unroll
  for (int i = 0; i < 8; ++i) z[i] = 0.0f;
  return z;
}

// ---------------------------------------------------------------------------
// Prep: swizzle W_a (H x 2H fp32, row-major) into bf16 WMMA B-fragments.
// Fragment (nt, kk): 32 lanes x 16 bf16 contiguous (1024 B). Lane L holds
// column n = nt*16 + (L&15); element e holds k = kk*32 + (e&7) + 16*(e>>3)
// + 8*(L>>4)  (mirrors the documented 16-bit A-fragment k striping).
// ---------------------------------------------------------------------------
__global__ void swizzle_wa(const float* __restrict__ W_a, __bf16* __restrict__ swz) {
  const int fid  = blockIdx.x * 8 + (threadIdx.x >> 5);   // 4032 fragments
  const int lane = threadIdx.x & 31;
  const int nt   = fid / KSTEPS;
  const int kk   = fid - nt * KSTEPS;
  const int n    = nt * 16 + (lane & 15);
  const int khi  = (lane >> 4) * 8;
  bf16x16 v;
#pragma unroll
  for (int e = 0; e < 16; ++e) {
    const int k = kk * 32 + (e & 7) + ((e >> 3) << 4) + khi;
    const float f = (n < HD && k < HD2) ? W_a[(size_t)n * HD2 + k] : 0.0f;
    v[e] = (__bf16)f;
  }
  *(bf16x16*)(swz + (size_t)fid * 512 + lane * 16) = v;
}

// hp_pad[b, h] = sum_d W_b[h, d] * decoder_hidden[0, b, d]   (zero padded to 1024)
__global__ void hp_kernel(const float* __restrict__ hidden, const float* __restrict__ W_b,
                          float* __restrict__ hp_pad) {
  const int b = blockIdx.y;
  const int h = blockIdx.x * 256 + threadIdx.x;
  if (h >= NPAD) return;
  float acc = 0.0f;
  if (h < HD) {
    const float* wr = W_b + (size_t)h * HD;
    const float* hv = hidden + (size_t)b * HD;
#pragma unroll 4
    for (int d = 0; d < HD; ++d) acc = fmaf(wr[d], hv[d], acc);
  }
  hp_pad[b * NPAD + h] = acc;
}

__global__ void wc_pad_kernel(const float* __restrict__ W_c_w, float* __restrict__ wc_pad) {
  const int n = blockIdx.x * 256 + threadIdx.x;
  if (n < NPAD) wc_pad[n] = (n < HD) ? W_c_w[n] : 0.0f;
}

// ---------------------------------------------------------------------------
// Scores: one WG per (s-block of 64, batch). 8 waves, 256 threads.
// LDS holds the 64 x 2016 bf16 A tile (row stride 2024 -> conflict-free).
// Wave w processes n-tile pairs {2p, 2p+1}, p = w + 8j, j = 0..3.
// score[b,s] = W_c_b + sum_n wc[n] * tanh(C[s,n] + hp[b,n])
// ---------------------------------------------------------------------------
__global__ __launch_bounds__(256, 1)
void scores_kernel(const float* __restrict__ ctx, const __bf16* __restrict__ swz,
                   const float* __restrict__ hp_pad, const float* __restrict__ wc_pad,
                   const float* __restrict__ W_c_b, float* __restrict__ scores) {
  extern __shared__ char smem[];
  __bf16* atile = (__bf16*)smem;                                  // MROWS * LDSROW bf16
  float*  sred  = (float*)(smem + (size_t)MROWS * LDSROW * 2);    // MROWS fp32

  const int b   = blockIdx.y;
  const int s0  = blockIdx.x * MROWS;
  const int tid = threadIdx.x;

  // ---- stage context rows into LDS as bf16 (zero-pad k to 2016) ----
  {
    const size_t base = ((size_t)b * SL + s0) * HD2;
    constexpr int CHUNKS = KPAD / 4;   // 504 float4 chunks per row
    for (int i = tid; i < MROWS * CHUNKS; i += 256) {
      const int row = i / CHUNKS;
      const int k4  = (i - row * CHUNKS) * 4;
      bf16x4 v4;
      if (k4 < HD2) {
        const f32x4 f = *(const f32x4*)(ctx + base + (size_t)row * HD2 + k4);
        v4[0] = (__bf16)f[0]; v4[1] = (__bf16)f[1];
        v4[2] = (__bf16)f[2]; v4[3] = (__bf16)f[3];
      } else {
        v4[0] = v4[1] = v4[2] = v4[3] = (__bf16)0.0f;
      }
      *(bf16x4*)(atile + (size_t)row * LDSROW + k4) = v4;
    }
  }
  if (tid < MROWS) sred[tid] = 0.0f;
  __syncthreads();

  const int w    = tid >> 5;
  const int lane = tid & 31;
  const int llo  = lane & 15;
  const int lhi  = lane >> 4;

  float spart[32];
#pragma unroll
  for (int i = 0; i < 32; ++i) spart[i] = 0.0f;

  for (int j = 0; j < 4; ++j) {
    const int pp  = w + 8 * j;
    const int nt0 = 2 * pp;

    f32x8 acc[4][2];
#pragma unroll
    for (int mt = 0; mt < 4; ++mt) { acc[mt][0] = zero8(); acc[mt][1] = zero8(); }

    const __bf16* b0 = swz + (size_t)nt0 * KSTEPS * 512 + lane * 16;
    const __bf16* b1 = b0 + (size_t)KSTEPS * 512;
    const __bf16* arow = atile + (size_t)llo * LDSROW + lhi * 8;

    for (int kk = 0; kk < KSTEPS; ++kk) {
      const bf16x16 bf0 = *(const bf16x16*)(b0 + kk * 512);
      const bf16x16 bf1 = *(const bf16x16*)(b1 + kk * 512);
      const __bf16* ap = arow + kk * 32;
#pragma unroll
      for (int mt = 0; mt < 4; ++mt) {
        const __bf16* r = ap + (size_t)mt * 16 * LDSROW;
        const bf16x8 lo = *(const bf16x8*)(r);        // k0 .. k0+7
        const bf16x8 hi = *(const bf16x8*)(r + 16);   // k0+16 .. k0+23
        const bf16x16 af = __builtin_shufflevector(
            lo, hi, 0, 1, 2, 3, 4, 5, 6, 7, 8, 9, 10, 11, 12, 13, 14, 15);
        acc[mt][0] = wmma_bf16(af, bf0, acc[mt][0]);
        acc[mt][1] = wmma_bf16(af, bf1, acc[mt][1]);
      }
    }

    // fused epilogue: wc[n] * tanh(C + hp[n]) folded into per-lane partials
#pragma unroll
    for (int u = 0; u < 2; ++u) {
      const int n  = (nt0 + u) * 16 + llo;
      const float wc = wc_pad[n];
      const float hv = hp_pad[b * NPAD + n];
#pragma unroll
      for (int mt = 0; mt < 4; ++mt)
#pragma unroll
        for (int v = 0; v < 8; ++v)
          spart[mt * 8 + v] += wc * tanhf(acc[mt][u][v] + hv);
    }
  }

  // reduce partials across lanes/waves: row m = mt*16 + v + 8*lhi
#pragma unroll
  for (int mt = 0; mt < 4; ++mt)
#pragma unroll
    for (int v = 0; v < 8; ++v)
      atomicAdd(&sred[mt * 16 + v + 8 * lhi], spart[mt * 8 + v]);
  __syncthreads();

  if (tid < MROWS)
    scores[(size_t)b * SL + s0 + tid] = sred[tid] + W_c_b[0];
}

// ---------------------------------------------------------------------------
// Finish: masked softmax over S, weighted sum over context, dense projection.
// One WG per batch.
// ---------------------------------------------------------------------------
__global__ __launch_bounds__(256)
void finish_kernel(const float* __restrict__ ctx, const unsigned char* __restrict__ mask,
                   const float* __restrict__ scores, const float* __restrict__ dense_w,
                   const float* __restrict__ dense_b, float* __restrict__ out) {
  __shared__ float p[SL];
  __shared__ float wsum[HD2];
  __shared__ float red[256];
  const int b = blockIdx.x;
  const int t = threadIdx.x;

  float s0 = scores[(size_t)b * SL + t];
  float s1 = scores[(size_t)b * SL + 256 + t];
  if (mask[b * SL + t])       s0 = -1e6f;
  if (mask[b * SL + 256 + t]) s1 = -1e6f;

  red[t] = fmaxf(s0, s1);
  __syncthreads();
  for (int off = 128; off > 0; off >>= 1) {
    if (t < off) red[t] = fmaxf(red[t], red[t + off]);
    __syncthreads();
  }
  const float mx = red[0];
  __syncthreads();

  const float e0 = __expf(s0 - mx);
  const float e1 = __expf(s1 - mx);
  red[t] = e0 + e1;
  __syncthreads();
  for (int off = 128; off > 0; off >>= 1) {
    if (t < off) red[t] += red[t + off];
    __syncthreads();
  }
  const float inv = 1.0f / red[0];
  p[t]       = e0 * inv;
  p[t + 256] = e1 * inv;
  __syncthreads();

  // weighted[d] = sum_s p[s] * context[b, s, d]   (coalesced over d)
  const float* cbase = ctx + (size_t)b * SL * HD2;
  for (int d = t; d < HD2; d += 256) {
    float acc = 0.0f;
    const float* cp = cbase + d;
#pragma unroll 4
    for (int s = 0; s < SL; ++s) acc = fmaf(p[s], cp[(size_t)s * HD2], acc);
    wsum[d] = acc;
  }
  __syncthreads();

  // out[b, h] = dense_b[h] + sum_d dense_w[h, d] * weighted[d]
  for (int h = t; h < HD; h += 256) {
    float acc = dense_b[h];
    const float* wp = dense_w + (size_t)h * HD2;
#pragma unroll 4
    for (int d = 0; d < HD2; ++d) acc = fmaf(wp[d], wsum[d], acc);
    out[(size_t)b * HD + h] = acc;
  }
}

// ---------------------------------------------------------------------------
// Launch
// ---------------------------------------------------------------------------
extern "C" void kernel_launch(void* const* d_in, const int* in_sizes, int n_in,
                              void* d_out, int out_size, void* d_ws, size_t ws_size,
                              hipStream_t stream) {
  const float*         hidden  = (const float*)d_in[0];          // (1, B, H)
  const float*         ctx     = (const float*)d_in[1];          // (B, S, 2H)
  const unsigned char* mask    = (const unsigned char*)d_in[2];  // (B, S, 1) bool
  const float*         W_a     = (const float*)d_in[3];          // (H, 2H)
  const float*         W_b     = (const float*)d_in[4];          // (H, H)
  const float*         W_c_w   = (const float*)d_in[5];          // (1, H)
  const float*         W_c_b   = (const float*)d_in[6];          // (1,)
  const float*         dense_w = (const float*)d_in[7];          // (H, 2H)
  const float*         dense_b = (const float*)d_in[8];          // (H,)
  float* out = (float*)d_out;

  char* ws = (char*)d_ws;
  const size_t swz_bytes = (size_t)NTILE * KSTEPS * 1024;        // 4,128,768
  __bf16* swz   = (__bf16*)(ws);
  float* hp_pad = (float*)(ws + swz_bytes);                      // 64*1024*4
  float* wc_pad = (float*)(ws + swz_bytes + (size_t)NB * NPAD * 4);
  float* scores = (float*)(ws + swz_bytes + (size_t)NB * NPAD * 4 + (size_t)NPAD * 4);

  swizzle_wa<<<dim3((NTILE * KSTEPS) / 8), dim3(256), 0, stream>>>(W_a, swz);
  hp_kernel<<<dim3(NPAD / 256, NB), dim3(256), 0, stream>>>(hidden, W_b, hp_pad);
  wc_pad_kernel<<<dim3(NPAD / 256), dim3(256), 0, stream>>>(W_c_w, wc_pad);

  const size_t ldsA = (size_t)MROWS * LDSROW * 2 + (size_t)MROWS * 4;  // ~259.3 KB
  scores_kernel<<<dim3(SL / MROWS, NB), dim3(256), ldsA, stream>>>(
      ctx, swz, hp_pad, wc_pad, W_c_b, scores);

  finish_kernel<<<dim3(NB), dim3(256), 0, stream>>>(ctx, mask, scores, dense_w, dense_b, out);
}
